// ASHModel_46445776339204
// MI455X (gfx1250) — compile-verified, matrix operand order checked by reading, not run on hardware
//
#include <hip/hip_runtime.h>
#include <hip/hip_bf16.h>

typedef _Float16 half_t;
typedef __attribute__((ext_vector_type(16))) _Float16 v16h;
typedef __attribute__((ext_vector_type(8)))  _Float16 v8h;
typedef __attribute__((ext_vector_type(8)))  float    v8f;

#define WMMA_F16(a, b, c) \
  __builtin_amdgcn_wmma_f32_16x16x32_f16(false, (a), false, (b), (short)0, (c), false, false)

// ---------------------------------------------------------------------------
// Elementwise helpers
// ---------------------------------------------------------------------------
__global__ void cast_f16_k(const float* __restrict__ src, half_t* __restrict__ dst, long n) {
  long i = blockIdx.x * (long)blockDim.x + threadIdx.x;
  if (i < n) dst[i] = (half_t)src[i];
}

// Pack conv weights OIHW f32 -> f16 [O][Kpad] rows (K zero-padded)
__global__ void pack_w_k(const float* __restrict__ src, half_t* __restrict__ dst,
                         int O, int Kreal, int Kpad) {
  long i = blockIdx.x * (long)blockDim.x + threadIdx.x;
  long n = (long)O * Kpad;
  if (i >= n) return;
  int o = (int)(i / Kpad), k = (int)(i - (long)o * Kpad);
  dst[i] = (k < Kreal) ? (half_t)src[(long)o * Kreal + k] : (half_t)0.f;
}

__global__ void fc_init_k(float* __restrict__ out, const float* __restrict__ bias,
                          int J, int JP) {
  int i = blockIdx.x * blockDim.x + threadIdx.x;
  int n = 32 * JP;
  if (i >= n) return;
  int j = i % JP;
  out[i] = (j < J) ? bias[j] : 0.f;
}

__global__ void relu_cast_k(const float* __restrict__ in, half_t* __restrict__ out, int n) {
  int i = blockIdx.x * blockDim.x + threadIdx.x;
  if (i >= n) return;
  float v = in[i];
  out[i] = (half_t)(v > 0.f ? v : 0.f);
}

// ---------------------------------------------------------------------------
// Implicit-GEMM conv + bias + relu, WMMA f16 -> f32 accum.
// M = B*H*W (spatial), N = Cout, K = Cin*9 (zero-padded to Kpad, mult of 32).
// Block: 256 threads = 8 wave32. 64x64 output tile, each wave owns 16x32.
// A staged via LDS (im2col gather w/ SAME padding).
// B (weights) staged via GLOBAL_LOAD_ASYNC_TO_LDS_B128: one 16B transfer per
// lane = 4KB tile, tracked on ASYNCcnt and overlapped with the A gather.
// ---------------------------------------------------------------------------
__global__ void __launch_bounds__(256)
conv_wmma_k(const half_t* __restrict__ in, const half_t* __restrict__ w,
            const float* __restrict__ bias, half_t* __restrict__ out,
            const int Cin, const int Cout, const int Hh, const int Ww,
            const int Kreal, const int Kpad) {
  __shared__ half_t ldsA[64 * 32];
  __shared__ half_t ldsB[64 * 32];

  const int t    = threadIdx.x;
  const int lane = t & 31;
  const int wave = t >> 5;
  const int mi   = wave & 3;   // 0..3  -> 16-row group
  const int nj   = wave >> 2;  // 0..1  -> 32-col group
  const int l16  = lane & 15;
  const int half = lane >> 4;

  const int mTile = blockIdx.x, nTile = blockIdx.y;
  const int HW = Hh * Ww;

  // --- cooperative A loader mapping: 2048 elems, 8 per thread ---
  const int ml = t >> 2;          // lds row 0..63
  const int kk = (t & 3) * 8;     // k offset within 32
  const int mg = mTile * 64 + ml;
  const int bb = mg / HW;
  const int rm0 = mg - bb * HW;
  const int yy = rm0 / Ww;
  const int xx = rm0 - yy * Ww;

  // --- cooperative B loader mapping: 64 rows x 64B, 16B per thread ---
  const int brow = t >> 2, bseg = t & 3;
  const half_t* wsrc = w + (size_t)(nTile * 64 + brow) * Kpad + bseg * 8;
  const unsigned ldsb_addr = (unsigned)(size_t)&ldsB[brow * 32 + bseg * 8];

  // --- WMMA fragment addressing ---
  const int arow = mi * 16 + l16;
  const int bcol0 = nj * 32 + l16;                 // local col for acc0
  const int n0 = nTile * 64 + bcol0;               // global col for acc0

  v8f acc0 = {}, acc1 = {};

  for (int k0 = 0; k0 < Kpad; k0 += 32) {
    // kick off async weight-tile DMA into LDS (ASYNCcnt)
    unsigned long long ga = (unsigned long long)(wsrc + k0);
    asm volatile("global_load_async_to_lds_b128 %0, %1, off"
                 :: "v"(ldsb_addr), "v"(ga) : "memory");

    // stage A tile (im2col gather, zero padded) while the DMA is in flight
    half_t tmp[8];
#pragma unroll
    for (int i = 0; i < 8; ++i) {
      int k = k0 + kk + i;
      half_t v = (half_t)0.f;
      if (k < Kreal) {
        int ci = k / 9, r9 = k - ci * 9;
        int kh = r9 / 3, kw = r9 - kh * 3;
        int iy = yy + kh - 1, ix = xx + kw - 1;
        if (iy >= 0 && iy < Hh && ix >= 0 && ix < Ww)
          v = in[(((size_t)bb * Cin + ci) * Hh + iy) * Ww + ix];
      }
      tmp[i] = v;
    }
    *(v8h*)&ldsA[ml * 32 + kk] = *(v8h*)tmp;

    // own async transfers done; barrier makes all waves' LDS writes visible
    asm volatile("s_wait_asynccnt 0x0" ::: "memory");
    __syncthreads();

    v16h a  = *(const v16h*)&ldsA[arow * 32 + half * 16];
    v16h b0 = *(const v16h*)&ldsB[bcol0 * 32 + half * 16];
    v16h b1 = *(const v16h*)&ldsB[(bcol0 + 16) * 32 + half * 16];
    acc0 = WMMA_F16(a, b0, acc0);
    acc1 = WMMA_F16(a, b1, acc1);
    __syncthreads();
  }

  // epilogue: bias + relu, write NCHW f16
  const float bi0 = bias[n0];
  const float bi1 = bias[n0 + 16];
#pragma unroll
  for (int r = 0; r < 8; ++r) {
    int mrow = mTile * 64 + mi * 16 + r + (lane < 16 ? 0 : 8);
    int b2 = mrow / HW;
    int rm = mrow - b2 * HW;
    int y2 = rm / Ww, x2 = rm - (rm / Ww) * Ww;
    float v0 = acc0[r] + bi0; v0 = v0 > 0.f ? v0 : 0.f;
    float v1 = acc1[r] + bi1; v1 = v1 > 0.f ? v1 : 0.f;
    out[(((size_t)b2 * Cout + n0)      * Hh + y2) * Ww + x2] = (half_t)v0;
    out[(((size_t)b2 * Cout + n0 + 16) * Hh + y2) * Ww + x2] = (half_t)v1;
  }
}

// ---------------------------------------------------------------------------
// Max pool p x p (stride p), f16 in/out, square maps
// ---------------------------------------------------------------------------
__global__ void pool_f16_k(const half_t* __restrict__ in, half_t* __restrict__ out,
                           int Cc, int Hin, int p, long n) {
  long i = blockIdx.x * (long)blockDim.x + threadIdx.x;
  if (i >= n) return;
  int Wo = Hin / p;
  int wo = (int)(i % Wo);
  long r = i / Wo;
  int ho = (int)(r % Wo); r /= Wo;
  int c  = (int)(r % Cc);
  int b  = (int)(r / Cc);
  float m = -1e30f;
  for (int dy = 0; dy < p; ++dy)
    for (int dx = 0; dx < p; ++dx) {
      float v = in[(((size_t)b * Cc + c) * Hin + ho * p + dy) * Hin + wo * p + dx];
      m = v > m ? v : m;
    }
  out[i] = (half_t)m;
}

// ---------------------------------------------------------------------------
// FC via WMMA, split-K with f32 atomics. Weights read as f32 from global and
// converted to f16 in registers (fc1 is pure-HBM bound: read weights once).
// blockDim = 128 (4 waves); wave ks index = blockIdx.z*4 + wave.
// Requires K == kChunk * gridDim.z * 4, kChunk % 32 == 0.
// ---------------------------------------------------------------------------
__global__ void __launch_bounds__(128)
fc_wmma_k(const float* __restrict__ Wt, const half_t* __restrict__ X,
          float* __restrict__ out, const int J, const int JP,
          const int Kdim, const int kChunk) {
  const int lane = threadIdx.x & 31;
  const int wave = threadIdx.x >> 5;
  const int jt = blockIdx.x, bt = blockIdx.y;
  const int ks = blockIdx.z * 4 + wave;
  const int k0beg = ks * kChunk, k0end = k0beg + kChunk;
  const int half = lane >> 4, l16 = lane & 15;

  int j  = jt * 16 + l16;
  int jc = j < J ? j : J - 1;
  const float*  ap = Wt + (size_t)jc * Kdim + half * 16;
  const half_t* bp = X  + (size_t)(bt * 16 + l16) * Kdim + half * 16;

  v8f acc = {};
  for (int k0 = k0beg; k0 < k0end; k0 += 32) {
    const float* ak = ap + k0;
    __builtin_prefetch(ak + 256, 0, 1);  // global_prefetch_b8 ahead of stream
    v16h a;
#pragma unroll
    for (int i = 0; i < 16; ++i) a[i] = (half_t)ak[i];
    v16h b = *(const v16h*)(bp + k0);
    acc = WMMA_F16(a, b, acc);
  }

#pragma unroll
  for (int r = 0; r < 8; ++r) {
    int jr = jt * 16 + r + (lane < 16 ? 0 : 8);
    if (jr < J) atomicAdd(&out[(size_t)(bt * 16 + l16) * JP + jr], acc[r]);
  }
}

// ---------------------------------------------------------------------------
// fc3 -> bbox (tiny: 32x4 dots of 512) + bbox offsets [-1,1,-1,1]
// ---------------------------------------------------------------------------
__global__ void fc3_bbox_k(const half_t* __restrict__ x, const float* __restrict__ w3,
                           const float* __restrict__ b3, float* __restrict__ bbox) {
  int t = threadIdx.x;
  if (t >= 128) return;
  int b = t >> 2, j = t & 3;
  float s = 0.f;
  for (int k = 0; k < 512; ++k) s += (float)x[b * 512 + k] * w3[j * 512 + k];
  const float offs[4] = {-1.f, 1.f, -1.f, 1.f};
  bbox[b * 4 + j] = s + b3[j] + offs[j];
}

// ---------------------------------------------------------------------------
// Gaussian glimpse: 32*768 threads, each computes one tuple element.
// l = yy*48 + xx*3 + cc ; writes f16 glimpse at [b][cc*256+yy*16+xx]
// ---------------------------------------------------------------------------
__global__ void glimpse_k(const float* __restrict__ img, const float* __restrict__ bbox,
                          const float* __restrict__ sigmas, const float* __restrict__ values,
                          const float* __restrict__ hbias, half_t* __restrict__ gout) {
  int tid = blockIdx.x * blockDim.x + threadIdx.x;
  if (tid >= 32 * 768) return;
  int b = tid / 768, l = tid % 768;
  int yy = l / 48, rem = l % 48, xx = rem / 3, cc = rem % 3;

  float ymin = bbox[b * 4 + 0], ymax = bbox[b * 4 + 1];
  float xmin = bbox[b * 4 + 2], xmax = bbox[b * 4 + 3];
  float py = yy * (1.f / 16.f), px = xx * (1.f / 16.f);
  float ry = py * (ymax - ymin) + ymin;
  float rx = px * (xmax - xmin) + xmin;
  float my = 127.f / (1.f + __expf(-ry));
  float mx = 127.f / (1.f + __expf(-rx));

  float sp  = sigmas[l] + 2.0f;
  float sof = sp > 20.f ? sp : log1pf(__expf(sp));
  float sig = (sof + 1e-7f) * 0.1f + 0.01f;
  float inv2s = 0.5f / (sig * sig);

  float fy = floorf(my), fx = floorf(mx);
  const float oy[4] = {0.f, 0.f, 1.f, 1.f};
  const float ox[4] = {0.f, 1.f, 0.f, 1.f};
  float dens[4], dsum = 0.f;
#pragma unroll
  for (int i = 0; i < 4; ++i) {
    float dy = fy + oy[i] - my, dx = fx + ox[i] - mx;
    dens[i] = __expf(-(dy * dy + dx * dx) * inv2s);
    dsum += dens[i];
  }
  float norm = 1.f / (dsum + 1e-7f);
  float tup = 0.f;
#pragma unroll
  for (int i = 0; i < 4; ++i) {
    int iy = (int)fminf(fmaxf(fy + oy[i], 0.f), 127.f);
    int ix = (int)fminf(fmaxf(fx + ox[i], 0.f), 127.f);
    float g = img[(((size_t)b * 3 + cc) * 128 + iy) * 128 + ix];
    tup += dens[i] * norm * g;
  }
  tup *= values[l];
  int flat = cc * 256 + yy * 16 + xx;
  gout[b * 768 + flat] = (half_t)(tup + hbias[flat]);
}

// ---------------------------------------------------------------------------
// Row softmax: 32 rows of 1000 (input stride 1008)
// ---------------------------------------------------------------------------
__global__ void softmax_k(const float* __restrict__ in, float* __restrict__ out) {
  __shared__ float red[256];
  int b = blockIdx.x, t = threadIdx.x;
  const float* row = in + (size_t)b * 1008;
  float m = -1e30f;
  for (int j = t; j < 1000; j += 256) m = fmaxf(m, row[j]);
  red[t] = m; __syncthreads();
  for (int s = 128; s > 0; s >>= 1) { if (t < s) red[t] = fmaxf(red[t], red[t + s]); __syncthreads(); }
  m = red[0]; __syncthreads();
  float sum = 0.f;
  for (int j = t; j < 1000; j += 256) sum += __expf(row[j] - m);
  red[t] = sum; __syncthreads();
  for (int s = 128; s > 0; s >>= 1) { if (t < s) red[t] += red[t + s]; __syncthreads(); }
  float inv = 1.f / red[0];
  for (int j = t; j < 1000; j += 256) out[(size_t)b * 1000 + j] = __expf(row[j] - m) * inv;
}

// ---------------------------------------------------------------------------
// Launch
// ---------------------------------------------------------------------------
extern "C" void kernel_launch(void* const* d_in, const int* in_sizes, int n_in,
                              void* d_out, int out_size, void* d_ws, size_t ws_size,
                              hipStream_t stream) {
  const float* image = (const float*)d_in[0];
  const float* cw1 = (const float*)d_in[1];  const float* cb1 = (const float*)d_in[2];
  const float* cw2 = (const float*)d_in[3];  const float* cb2 = (const float*)d_in[4];
  const float* cw3 = (const float*)d_in[5];  const float* cb3 = (const float*)d_in[6];
  const float* cw4 = (const float*)d_in[7];  const float* cb4 = (const float*)d_in[8];
  const float* cw5 = (const float*)d_in[9];  const float* cb5 = (const float*)d_in[10];
  const float* cw6 = (const float*)d_in[11]; const float* cb6 = (const float*)d_in[12];
  const float* fw1 = (const float*)d_in[13]; const float* fb1 = (const float*)d_in[14];
  const float* fw2 = (const float*)d_in[15]; const float* fb2 = (const float*)d_in[16];
  const float* fw3 = (const float*)d_in[17]; const float* fb3 = (const float*)d_in[18];
  const float* sigmas = (const float*)d_in[19];
  const float* values = (const float*)d_in[20];
  const float* hbias  = (const float*)d_in[21];
  const float* l1w = (const float*)d_in[22]; const float* l1b = (const float*)d_in[23];
  const float* l2w = (const float*)d_in[24]; const float* l2b = (const float*)d_in[25];
  float* outp = (float*)d_out;

  char* ws = (char*)d_ws;
  size_t off = 0;
  auto alloc = [&](size_t bytes) -> char* {
    char* p = ws + off;
    off += (bytes + 255) & ~(size_t)255;
    return p;
  };

  half_t* img16 = (half_t*)alloc((size_t)32 * 3 * 128 * 128 * 2);
  half_t* actA  = (half_t*)alloc((size_t)32 * 64 * 128 * 128 * 2);
  half_t* actB  = (half_t*)alloc((size_t)32 * 64 * 128 * 128 * 2);
  half_t* w1p = (half_t*)alloc((size_t)64 * 32 * 2);
  half_t* w2p = (half_t*)alloc((size_t)64 * 576 * 2);
  half_t* w3p = (half_t*)alloc((size_t)128 * 576 * 2);
  half_t* w4p = (half_t*)alloc((size_t)128 * 1152 * 2);
  half_t* w5p = (half_t*)alloc((size_t)512 * 1152 * 2);
  half_t* w6p = (half_t*)alloc((size_t)512 * 4608 * 2);
  float*  fc1o = (float*)alloc((size_t)32 * 512 * 4);
  half_t* x1   = (half_t*)alloc((size_t)32 * 512 * 2);
  float*  fc2o = (float*)alloc((size_t)32 * 512 * 4);
  half_t* x2   = (half_t*)alloc((size_t)32 * 512 * 2);
  float*  bbox = (float*)alloc((size_t)32 * 4 * 4);
  half_t* g16  = (half_t*)alloc((size_t)32 * 768 * 2);
  float*  l1o  = (float*)alloc((size_t)32 * 512 * 4);
  half_t* h16  = (half_t*)alloc((size_t)32 * 512 * 2);
  float*  l2o  = (float*)alloc((size_t)32 * 1008 * 4);

  // --- weight packing + input cast ---
  long nimg = (long)32 * 3 * 128 * 128;
  cast_f16_k<<<(nimg + 255) / 256, 256, 0, stream>>>(image, img16, nimg);
  pack_w_k<<<(64 * 32 + 255) / 256, 256, 0, stream>>>(cw1, w1p, 64, 27, 32);
  pack_w_k<<<(64 * 576 + 255) / 256, 256, 0, stream>>>(cw2, w2p, 64, 576, 576);
  pack_w_k<<<(128 * 576 + 255) / 256, 256, 0, stream>>>(cw3, w3p, 128, 576, 576);
  pack_w_k<<<(128 * 1152 + 255) / 256, 256, 0, stream>>>(cw4, w4p, 128, 1152, 1152);
  pack_w_k<<<(512 * 1152 + 255) / 256, 256, 0, stream>>>(cw5, w5p, 512, 1152, 1152);
  pack_w_k<<<(512 * 4608 + 255) / 256, 256, 0, stream>>>(cw6, w6p, 512, 4608, 4608);

  // --- conv stack (WMMA implicit GEMM, async weight staging) ---
  conv_wmma_k<<<dim3(8192, 1), 256, 0, stream>>>(img16, w1p, cb1, actA, 3, 64, 128, 128, 27, 32);
  conv_wmma_k<<<dim3(8192, 1), 256, 0, stream>>>(actA, w2p, cb2, actB, 64, 64, 128, 128, 576, 576);
  long np1 = (long)32 * 64 * 32 * 32;
  pool_f16_k<<<(np1 + 255) / 256, 256, 0, stream>>>(actB, actA, 64, 128, 4, np1);
  conv_wmma_k<<<dim3(512, 2), 256, 0, stream>>>(actA, w3p, cb3, actB, 64, 128, 32, 32, 576, 576);
  conv_wmma_k<<<dim3(512, 2), 256, 0, stream>>>(actB, w4p, cb4, actA, 128, 128, 32, 32, 1152, 1152);
  long np2 = (long)32 * 128 * 16 * 16;
  pool_f16_k<<<(np2 + 255) / 256, 256, 0, stream>>>(actA, actB, 128, 32, 2, np2);
  conv_wmma_k<<<dim3(128, 8), 256, 0, stream>>>(actB, w5p, cb5, actA, 128, 512, 16, 16, 1152, 1152);
  conv_wmma_k<<<dim3(128, 8), 256, 0, stream>>>(actA, w6p, cb6, actB, 512, 512, 16, 16, 4608, 4608);
  // actB now holds conv6 output = fc1 input, f16 [32][131072]

  // --- fc1 (HBM-bound split-K WMMA over f32 weights) ---
  fc_init_k<<<(32 * 512 + 255) / 256, 256, 0, stream>>>(fc1o, fb1, 512, 512);
  fc_wmma_k<<<dim3(32, 2, 64), 128, 0, stream>>>(fw1, actB, fc1o, 512, 512, 131072, 512);
  relu_cast_k<<<(32 * 512 + 255) / 256, 256, 0, stream>>>(fc1o, x1, 32 * 512);

  // --- fc2 ---
  fc_init_k<<<(32 * 512 + 255) / 256, 256, 0, stream>>>(fc2o, fb2, 512, 512);
  fc_wmma_k<<<dim3(32, 2, 1), 128, 0, stream>>>(fw2, x1, fc2o, 512, 512, 512, 128);
  relu_cast_k<<<(32 * 512 + 255) / 256, 256, 0, stream>>>(fc2o, x2, 32 * 512);

  // --- fc3 -> bbox, glimpse ---
  fc3_bbox_k<<<1, 128, 0, stream>>>(x2, fw3, fb3, bbox);
  glimpse_k<<<(32 * 768 + 255) / 256, 256, 0, stream>>>(image, bbox, sigmas, values, hbias, g16);

  // --- l1 ---
  fc_init_k<<<(32 * 512 + 255) / 256, 256, 0, stream>>>(l1o, l1b, 512, 512);
  fc_wmma_k<<<dim3(32, 2, 1), 128, 0, stream>>>(l1w, g16, l1o, 512, 512, 768, 192);
  relu_cast_k<<<(32 * 512 + 255) / 256, 256, 0, stream>>>(l1o, h16, 32 * 512);

  // --- l2 (J=1000 padded to 1008) + softmax ---
  fc_init_k<<<(32 * 1008 + 255) / 256, 256, 0, stream>>>(l2o, l2b, 1000, 1008);
  fc_wmma_k<<<dim3(63, 2, 1), 128, 0, stream>>>(l2w, h16, l2o, 1000, 1008, 512, 128);
  softmax_k<<<32, 256, 0, stream>>>(l2o, outp);
}
